// Pool_2224793059944
// MI455X (gfx1250) — 8-bit matrix operand order repaired and checked by reading, compiled, not run on hardware
//
#include <hip/hip_runtime.h>
#include <hip/hip_bf16.h>
#include <cstdint>

#define NN 4096
#define DD 512
#define KK 2048

typedef int v8i __attribute__((ext_vector_type(8)));

__global__ __launch_bounds__(256)
void scores_kernel(const float* __restrict__ h, const float* __restrict__ W,
                   const float* __restrict__ b, float* __restrict__ scores) {
    int gid  = blockIdx.x * blockDim.x + threadIdx.x;
    int wave = gid >> 5;
    int lane = threadIdx.x & 31;
    if (wave >= NN) return;
    const float* hr = h + (size_t)wave * DD;
    float s = 0.f;
    for (int d = lane; d < DD; d += 32) s += hr[d] * W[d];
    for (int off = 16; off > 0; off >>= 1) s += __shfl_down(s, off, 32);
    if (lane == 0) scores[wave] = 1.f / (1.f + __expf(-(s + b[0])));
}

__global__ __launch_bounds__(1024)
void topk_sort_kernel(const float* __restrict__ scores,
                      int* __restrict__ sidx, float* __restrict__ svals) {
    __shared__ float sk[NN];
    __shared__ int   sv[NN];
    for (int i = threadIdx.x; i < NN; i += 1024) { sk[i] = scores[i]; sv[i] = i; }
    __syncthreads();
    for (int k = 2; k <= NN; k <<= 1) {
        for (int j = k >> 1; j > 0; j >>= 1) {
            for (int i = threadIdx.x; i < NN; i += 1024) {
                int ixj = i ^ j;
                if (ixj > i) {
                    float ka = sk[i], kb = sk[ixj];
                    int   va = sv[i], vb = sv[ixj];
                    bool descSeg  = ((i & k) == 0);
                    bool aBeforeB = (ka > kb) || (ka == kb && va < vb);
                    bool doSwap   = descSeg ? (!aBeforeB) : aBeforeB;
                    if (doSwap) { sk[i]=kb; sk[ixj]=ka; sv[i]=vb; sv[ixj]=va; }
                }
            }
            __syncthreads();
        }
    }
    for (int i = threadIdx.x; i < KK; i += 1024) { sidx[i] = sv[i]; svals[i] = sk[i]; }
}

__global__ __launch_bounds__(256)
void packA_kernel(const float* __restrict__ g, const int* __restrict__ sidx,
                  unsigned char* __restrict__ A8) {
    int t  = blockIdx.x * blockDim.x + threadIdx.x;
    int r  = t >> 10;
    int k4 = (t & 1023) << 2;
    if (r >= KK) return;
    const float4 f = *(const float4*)(g + (size_t)sidx[r] * NN + k4);
    unsigned int pk = (f.x != 0.f ? 1u       : 0u) | (f.y != 0.f ? 1u << 8  : 0u)
                    | (f.z != 0.f ? 1u << 16 : 0u) | (f.w != 0.f ? 1u << 24 : 0u);
    *(unsigned int*)(A8 + (size_t)r * NN + k4) = pk;
}

__global__ __launch_bounds__(256)
void packB_kernel(const float* __restrict__ g, const int* __restrict__ sidx,
                  unsigned char* __restrict__ B8) {
    int t  = blockIdx.x * blockDim.x + threadIdx.x;
    int c  = t >> 10;
    int k4 = (t & 1023) << 2;
    if (c >= KK) return;
    int j = sidx[c];
    unsigned int pk = 0;
    #pragma unroll
    for (int i = 0; i < 4; ++i) {
        float v = g[(size_t)(k4 + i) * NN + j];
        pk |= (v != 0.f ? 1u : 0u) << (8 * i);
    }
    *(unsigned int*)(B8 + (size_t)c * NN + k4) = pk;
}

__global__ __launch_bounds__(256)
void gemm2hop_kernel(const unsigned char* __restrict__ A8,
                     const unsigned char* __restrict__ B8,
                     unsigned char* __restrict__ G2) {
    const int lane  = threadIdx.x & 31;
    const int wave  = threadIdx.x >> 5;
    const int waveM = wave & 1;
    const int waveN = wave >> 1;
    const int Mbase = blockIdx.y * 128 + waveM * 64;
    const int Nbase = blockIdx.x * 256 + waveN * 64;
    const int lmod  = lane & 15;
    const int lhalf = lane >> 4;

    v8i acc[4][4] = {};

    for (int kb = 0; kb < NN; kb += 64) {
        v8i a[4], b[4];
        #pragma unroll
        for (int mi = 0; mi < 4; ++mi) {
            const unsigned char* ap =
                A8 + (size_t)(Mbase + mi * 16 + lmod) * NN + kb + lhalf * 16;
            int2 p0 = *(const int2*)(ap +  0);
            int2 p1 = *(const int2*)(ap +  8);
            int2 p2 = *(const int2*)(ap + 32);
            int2 p3 = *(const int2*)(ap + 40);
            a[mi][0] = p0.x; a[mi][1] = p0.y;
            a[mi][2] = p1.x; a[mi][3] = p1.y;
            a[mi][4] = p2.x; a[mi][5] = p2.y;
            a[mi][6] = p3.x; a[mi][7] = p3.y;
            __builtin_prefetch(ap + 64, 0, 3);
        }
        #pragma unroll
        for (int ni = 0; ni < 4; ++ni) {
            const unsigned char* bp =
                B8 + (size_t)(Nbase + ni * 16 + lmod) * NN + kb + lhalf * 16;
            int4 q0 = *(const int4*)(bp +  0);
            int4 q1 = *(const int4*)(bp + 32);
            b[ni][0] = q0.x; b[ni][1] = q0.y; b[ni][2] = q0.z; b[ni][3] = q0.w;
            b[ni][4] = q1.x; b[ni][5] = q1.y; b[ni][6] = q1.z; b[ni][7] = q1.w;
            __builtin_prefetch(bp + 64, 0, 3);
        }
        #pragma unroll
        for (int mi = 0; mi < 4; ++mi)
            #pragma unroll
            for (int ni = 0; ni < 4; ++ni)
                acc[mi][ni] = __builtin_amdgcn_wmma_i32_16x16x64_iu8(
                    false, a[mi], false, b[ni], acc[mi][ni], false, false);
    }

    #pragma unroll
    for (int mi = 0; mi < 4; ++mi)
        #pragma unroll
        for (int ni = 0; ni < 4; ++ni)
            #pragma unroll
            for (int v = 0; v < 8; ++v) {
                int r = Mbase + mi * 16 + lhalf * 8 + v;
                int c = Nbase + ni * 16 + lmod;
                G2[(size_t)r * KK + c] = (acc[mi][ni][v] != 0) ? (unsigned char)1
                                                               : (unsigned char)0;
            }
}

__global__ __launch_bounds__(256)
void dinv_kernel(const unsigned char* __restrict__ G2, float* __restrict__ dinv) {
    int wave = (blockIdx.x * blockDim.x + threadIdx.x) >> 5;
    int lane = threadIdx.x & 31;
    if (wave >= KK) return;
    const unsigned int* row = (const unsigned int*)(G2 + (size_t)wave * KK);
    unsigned int s = 0;
    for (int k = lane; k < KK / 4; k += 32) {
        unsigned int w = row[k];
        s += (w & 0xFFu) + ((w >> 8) & 0xFFu) + ((w >> 16) & 0xFFu) + ((w >> 24) & 0xFFu);
    }
    for (int off = 16; off > 0; off >>= 1) s += __shfl_down(s, off, 32);
    if (lane == 0) dinv[wave] = rsqrtf((float)s);
}

__global__ __launch_bounds__(256)
void norm_kernel(const unsigned char* __restrict__ G2,
                 const float* __restrict__ dinv, float* __restrict__ out) {
    int t  = blockIdx.x * blockDim.x + threadIdx.x;
    int r  = t >> 9;
    int c4 = (t & 511) << 2;
    if (r >= KK) return;
    unsigned int w = *(const unsigned int*)(G2 + (size_t)r * KK + c4);
    float dr = dinv[r];
    float4 o;
    o.x = ((w      ) & 0xFFu) ? dr * dinv[c4 + 0] : 0.f;
    o.y = ((w >>  8) & 0xFFu) ? dr * dinv[c4 + 1] : 0.f;
    o.z = ((w >> 16) & 0xFFu) ? dr * dinv[c4 + 2] : 0.f;
    o.w = ((w >> 24) & 0xFFu) ? dr * dinv[c4 + 3] : 0.f;
    if (r - c4 >= 0 && r - c4 < 4) {
        if (r == c4 + 0) o.x += 1.f;
        if (r == c4 + 1) o.y += 1.f;
        if (r == c4 + 2) o.z += 1.f;
        if (r == c4 + 3) o.w += 1.f;
    }
    *(float4*)(out + (size_t)r * KK + c4) = o;
}

__global__ __launch_bounds__(256)
void newh_kernel(const float* __restrict__ h, const int* __restrict__ sidx,
                 const float* __restrict__ svals, float* __restrict__ outH) {
    int t  = blockIdx.x * blockDim.x + threadIdx.x;
    int r  = t >> 7;
    int d4 = (t & 127) << 2;
    if (r >= KK) return;
    float4 v = *(const float4*)(h + (size_t)sidx[r] * DD + d4);
    float s = svals[r];
    v.x *= s; v.y *= s; v.z *= s; v.w *= s;
    *(float4*)(outH + (size_t)r * DD + d4) = v;
}

__global__ __launch_bounds__(256)
void idxout_kernel(const int* __restrict__ sidx, float* __restrict__ outI) {
    int t = blockIdx.x * blockDim.x + threadIdx.x;
    if (t < KK) outI[t] = (float)sidx[t];
}

extern "C" void kernel_launch(void* const* d_in, const int* in_sizes, int n_in,
                              void* d_out, int out_size, void* d_ws, size_t ws_size,
                              hipStream_t stream) {
    const float* g = (const float*)d_in[0];
    const float* h = (const float*)d_in[1];
    const float* W = (const float*)d_in[2];
    const float* b = (const float*)d_in[3];
    float* out = (float*)d_out;

    char* ws = (char*)d_ws;
    float*         scores = (float*)(ws + 0);
    int*           sidx   = (int*)  (ws + 16384);
    float*         svals  = (float*)(ws + 32768);
    float*         dinv   = (float*)(ws + 49152);
    unsigned char* A8     = (unsigned char*)(ws + 57344);
    unsigned char* B8     = A8 + (size_t)KK * NN;
    unsigned char* G2     = B8 + (size_t)KK * NN;

    scores_kernel   <<<512, 256, 0, stream>>>(h, W, b, scores);
    topk_sort_kernel<<<1, 1024, 0, stream>>>(scores, sidx, svals);
    packA_kernel    <<<(KK * 1024) / 256, 256, 0, stream>>>(g, sidx, A8);
    packB_kernel    <<<(KK * 1024) / 256, 256, 0, stream>>>(g, sidx, B8);
    dim3 gg(KK / 256, KK / 128);
    gemm2hop_kernel <<<gg, 256, 0, stream>>>(A8, B8, G2);
    dinv_kernel     <<<KK / 8, 256, 0, stream>>>(G2, dinv);
    norm_kernel     <<<(KK * KK / 4) / 256, 256, 0, stream>>>(G2, dinv, out);
    newh_kernel     <<<(KK * DD / 4) / 256, 256, 0, stream>>>(h, sidx, svals,
                                                              out + (size_t)KK * KK);
    idxout_kernel   <<<(KK + 255) / 256, 256, 0, stream>>>(sidx,
                                         out + (size_t)KK * KK + (size_t)KK * DD);
}
